// NOCD_DL_59536836657814
// MI455X (gfx1250) — compile-verified, matrix operand order checked by reading, not run on hardware
//
#include <hip/hip_runtime.h>
#include <hip/hip_bf16.h>
#include <math.h>
#include <stdint.h>

typedef __attribute__((ext_vector_type(16))) __bf16 v16bf;
typedef __attribute__((ext_vector_type(8)))  float  v8f;

#define GEMM_BM 128
#define GEMM_BN 64
#define GEMM_BK 32
#define A_STRIDE 36   // floats per LDS A row: 144B (16B aligned) + bank spread
#define B_STRIDE 68   // floats per LDS B row: 272B (16B aligned)

// ---------------------------------------------------------------------------
// Wave-level reduction helpers (wave32 on gfx1250)
// ---------------------------------------------------------------------------
__device__ inline float wave_sum(float v) {
#pragma unroll
  for (int m = 16; m > 0; m >>= 1) v += __shfl_xor(v, m, 32);
  return v;
}
__device__ inline float wave_max(float v) {
#pragma unroll
  for (int m = 16; m > 0; m >>= 1) v = fmaxf(v, __shfl_xor(v, m, 32));
  return v;
}

// ---------------------------------------------------------------------------
// WMMA GEMM: C[M,Ncols] = act( rowScale.A[M,K] @ B[K,Ncols] + bias + (acc?C:0) )
// Block = 256 threads = 8 waves; block tile 128x64; each wave owns 32x32
// (2x2 v_wmma_f32_16x16x32_bf16 tiles). Interior tiles are staged into LDS
// with global_load_async_to_lds_b128 (ASYNCcnt); edge tiles use branchless
// clamped loads. LDS holds fp32; gather converts to bf16 and applies rowScale.
// ---------------------------------------------------------------------------
__global__ __launch_bounds__(256)
void gemm_wmma_bf16(const float* __restrict__ A, const float* __restrict__ B,
                    const float* __restrict__ bias,
                    const float* __restrict__ rowScale, int rsStride,
                    float* __restrict__ C,
                    int M, int K, int Ncols, int act, int accumulate)
{
  __shared__ float As[GEMM_BM * A_STRIDE];   // [row][k]
  __shared__ float Bs[GEMM_BK * B_STRIDE];   // [k][n]

  const int tid  = threadIdx.x;
  const int lane = tid & 31;
  const int wave = tid >> 5;
  const int wm   = wave >> 1;          // 0..3 : 32-row band
  const int wn   = wave & 1;           // 0..1 : 32-col band
  const int half = lane >> 4;          // ISA 16-bit layout K-half selector
  const int mr   = lane & 15;
  const int khalf = 8 * half;
  const int blockM = blockIdx.y * GEMM_BM;
  const int blockN = blockIdx.x * GEMM_BN;

  const bool fullM = (blockM + GEMM_BM <= M);
  const bool fullN = (blockN + GEMM_BN <= Ncols);

  // per-lane row scales for the two A subtiles (uniform per output row)
  float rsc[2] = {1.f, 1.f};
  if (rowScale) {
#pragma unroll
    for (int mi = 0; mi < 2; ++mi) {
      int row = blockM + wm * 32 + mi * 16 + mr;
      row = row < M ? row : M - 1;
      rsc[mi] = rowScale[(size_t)row * rsStride];
    }
  }

  v8f acc[2][2] = {};
  if (accumulate) {
#pragma unroll
    for (int mi = 0; mi < 2; ++mi)
#pragma unroll
      for (int ni = 0; ni < 2; ++ni)
#pragma unroll
        for (int v = 0; v < 8; ++v) {
          int row = blockM + wm * 32 + mi * 16 + v + khalf;
          int col = blockN + wn * 32 + ni * 16 + mr;
          if (row < M && col < Ncols)
            acc[mi][ni][v] = C[(size_t)row * Ncols + col];
        }
  }

  for (int k0 = 0; k0 < K; k0 += GEMM_BK) {
    const bool fullK = (k0 + GEMM_BK <= K);

    // ---- stage A tile: 128x32 fp32 ----
    if (fullM && fullK) {
#pragma unroll
      for (int it = 0; it < 4; ++it) {
        int idx = it * 256 + tid;              // 1024 x 16B chunks
        int r  = idx >> 3;
        int c4 = (idx & 7) << 2;
        const float* g = A + (size_t)(blockM + r) * K + (k0 + c4);
        unsigned l = (unsigned)(uintptr_t)&As[r * A_STRIDE + c4];
        asm volatile("global_load_async_to_lds_b128 %0, %1, off"
                     :: "v"(l), "v"(g) : "memory");
      }
      if (k0 + 2 * GEMM_BK <= K) {             // prefetch next K-chunk
        const float* g = A + (size_t)(blockM + (tid >> 3)) * K +
                         (k0 + GEMM_BK + ((tid & 7) << 2));
        __builtin_prefetch(g, 0, 1);
      }
    } else {
      for (int i = tid; i < GEMM_BM * GEMM_BK; i += 256) {
        int r = i >> 5, c = i & 31;
        int gr = blockM + r, gk = k0 + c;
        bool in = (gr < M) && (gk < K);
        int grc = gr < M ? gr : M - 1;
        int gkc = gk < K ? gk : K - 1;
        float v = A[(size_t)grc * K + gkc];
        As[r * A_STRIDE + c] = in ? v : 0.f;
      }
    }

    // ---- stage B tile: 32x64 fp32, kept [k][n] (memory order) ----
    if (fullN && fullK) {
#pragma unroll
      for (int it = 0; it < 2; ++it) {
        int idx = it * 256 + tid;              // 512 x 16B chunks
        int k  = idx >> 4;
        int n4 = (idx & 15) << 2;
        const float* g = B + (size_t)(k0 + k) * Ncols + (blockN + n4);
        unsigned l = (unsigned)(uintptr_t)&Bs[k * B_STRIDE + n4];
        asm volatile("global_load_async_to_lds_b128 %0, %1, off"
                     :: "v"(l), "v"(g) : "memory");
      }
    } else {
      for (int i = tid; i < GEMM_BK * GEMM_BN; i += 256) {
        int k = i >> 6, n = i & 63;
        int gk = k0 + k, gn = blockN + n;
        bool in = (gk < K) && (gn < Ncols);
        int gkc = gk < K ? gk : K - 1;
        int gnc = gn < Ncols ? gn : Ncols - 1;
        float v = B[(size_t)gkc * Ncols + gnc];
        Bs[k * B_STRIDE + n] = in ? v : 0.f;
      }
    }

    if ((fullM || fullN) && fullK)
      asm volatile("s_wait_asynccnt 0x0" ::: "memory");
    __syncthreads();

    // ---- gather fragments (ISA 7.12.2 wave32 16-bit layouts), cvt f32->bf16
    v16bf af[2], bf[2];
#pragma unroll
    for (int p = 0; p < 8; ++p) {
      int kb = ((p < 4) ? (2 * p) : (16 + 2 * (p - 4))) + khalf;
#pragma unroll
      for (int mi = 0; mi < 2; ++mi) {
        const float* ap = &As[(wm * 32 + mi * 16 + mr) * A_STRIDE];
        af[mi][2 * p]     = (__bf16)(ap[kb]     * rsc[mi]);
        af[mi][2 * p + 1] = (__bf16)(ap[kb + 1] * rsc[mi]);
      }
#pragma unroll
      for (int ni = 0; ni < 2; ++ni) {
        int col = wn * 32 + ni * 16 + mr;
        bf[ni][2 * p]     = (__bf16)Bs[kb * B_STRIDE + col];
        bf[ni][2 * p + 1] = (__bf16)Bs[(kb + 1) * B_STRIDE + col];
      }
    }
#pragma unroll
    for (int mi = 0; mi < 2; ++mi)
#pragma unroll
      for (int ni = 0; ni < 2; ++ni)
        acc[mi][ni] = __builtin_amdgcn_wmma_f32_16x16x32_bf16(
            false, af[mi], false, bf[ni], (short)0, acc[mi][ni], false, false);
    __syncthreads();
  }

  // ---- epilogue: bias + optional ELU, guarded stores ----
#pragma unroll
  for (int mi = 0; mi < 2; ++mi)
#pragma unroll
    for (int ni = 0; ni < 2; ++ni)
#pragma unroll
      for (int v = 0; v < 8; ++v) {
        int row = blockM + wm * 32 + mi * 16 + v + khalf;
        int col = blockN + wn * 32 + ni * 16 + mr;
        if (row < M && col < Ncols) {
          float val = acc[mi][ni][v];
          if (bias) val += bias[col];
          if (act == 1) val = (val >= 0.f) ? val : expm1f(val);   // ELU
          C[(size_t)row * Ncols + col] = val;
        }
      }
}

// ---------------------------------------------------------------------------
__global__ void fill_zero(float* __restrict__ p, size_t n) {
  size_t i = (size_t)blockIdx.x * blockDim.x + threadIdx.x;
  if (i < n) p[i] = 0.f;
}

// COO SpMM: Out[row[e]] += val[e] * H[col[e]]  (one block per edge)
__global__ __launch_bounds__(256)
void spmm_kernel(const int* __restrict__ rows, const int* __restrict__ cols,
                 const float* __restrict__ vals, const float* __restrict__ H,
                 float* __restrict__ Out, int E, int D) {
  int e = blockIdx.x;
  if (e >= E) return;
  int r = rows[e], c = cols[e];
  float v = vals[e];
  const float* hs = H + (size_t)c * D;
  float* os = Out + (size_t)r * D;
  for (int d = threadIdx.x; d < D; d += blockDim.x)
    atomicAdd(os + d, v * hs[d]);
}

// elementwise: optional leaky_relu(0.2) then ELU, in place
__global__ void act_kernel(float* __restrict__ X, size_t n, int lrelu) {
  size_t i = (size_t)blockIdx.x * blockDim.x + threadIdx.x;
  if (i >= n) return;
  float x = X[i];
  if (lrelu && x < 0.f) x *= 0.2f;
  X[i] = (x >= 0.f) ? x : expm1f(x);
}

// BatchNorm (affine=False, batch stats, biased var): column stats pass
__global__ __launch_bounds__(256)
void bn_stats_kernel(const float* __restrict__ X, int N, int D,
                     float* __restrict__ stats) {
  __shared__ float ssum[8][33];
  __shared__ float ssq[8][33];
  int tx = threadIdx.x;           // column within group of 32
  int ty = threadIdx.y;           // row-stride lane
  int c = blockIdx.x * 32 + tx;
  float s = 0.f, q = 0.f;
  if (c < D) {
    for (int r = ty; r < N; r += 8) {
      float v = X[(size_t)r * D + c];
      s += v; q += v * v;
    }
  }
  ssum[ty][tx] = s; ssq[ty][tx] = q;
  __syncthreads();
  if (ty == 0 && c < D) {
    float ts = 0.f, tq = 0.f;
#pragma unroll
    for (int j = 0; j < 8; ++j) { ts += ssum[j][tx]; tq += ssq[j][tx]; }
    float mean = ts / (float)N;
    float var  = tq / (float)N - mean * mean;
    stats[c]        = mean;
    stats[2048 + c] = rsqrtf(var + 1e-5f);
  }
}

__global__ void bn_apply_kernel(const float* __restrict__ X, float* __restrict__ Y,
                                int N, int D, const float* __restrict__ stats) {
  size_t i = (size_t)blockIdx.x * blockDim.x + threadIdx.x;
  if (i >= (size_t)N * D) return;
  int c = (int)(i % (size_t)D);
  Y[i] = (X[i] - stats[c]) * stats[2048 + c];
}

// Gate u = F.normalize(softmax(tanh(coms @ mlpl_w + b)))  -- one wave per row,
// coms gathered piecewise from t1..t4 (no 123MB concat buffer).
__global__ __launch_bounds__(256)
void u_kernel(const float* __restrict__ T1, const float* __restrict__ T2,
              const float* __restrict__ T3, const float* __restrict__ T4,
              const float* __restrict__ W, const float* __restrict__ Bv,
              float* __restrict__ U, int N) {
  int w = blockIdx.x * 8 + (threadIdx.x >> 5);
  int lane = threadIdx.x & 31;
  if (w >= N) return;
  float acc[5] = {0.f, 0.f, 0.f, 0.f, 0.f};
  const float* t1 = T1 + (size_t)w * 500;
  const float* t2 = T2 + (size_t)w * 500;
  const float* t3 = T3 + (size_t)w * 2000;
  const float* t4 = T4 + (size_t)w * 64;
  for (int k = lane; k < 500; k += 32) {
    float a = t1[k]; const float* wr = W + (size_t)k * 5;
#pragma unroll
    for (int j = 0; j < 5; ++j) acc[j] += a * wr[j];
  }
  for (int k = lane; k < 500; k += 32) {
    float a = t2[k]; const float* wr = W + (size_t)(500 + k) * 5;
#pragma unroll
    for (int j = 0; j < 5; ++j) acc[j] += a * wr[j];
  }
  for (int k = lane; k < 2000; k += 32) {
    float a = t3[k]; const float* wr = W + (size_t)(1000 + k) * 5;
#pragma unroll
    for (int j = 0; j < 5; ++j) acc[j] += a * wr[j];
  }
  for (int k = lane; k < 64; k += 32) {
    float a = t4[k]; const float* wr = W + (size_t)(3000 + k) * 5;
#pragma unroll
    for (int j = 0; j < 5; ++j) acc[j] += a * wr[j];
  }
#pragma unroll
  for (int j = 0; j < 5; ++j) acc[j] = wave_sum(acc[j]);
  if (lane == 0) {
    float t[5]; float mx = -1e30f;
#pragma unroll
    for (int j = 0; j < 5; ++j) { t[j] = tanhf(acc[j] + Bv[j]); mx = fmaxf(mx, t[j]); }
    float ssm = 0.f;
#pragma unroll
    for (int j = 0; j < 5; ++j) { t[j] = expf(t[j] - mx); ssm += t[j]; }
    float n2 = 0.f;
#pragma unroll
    for (int j = 0; j < 5; ++j) { t[j] /= ssm; n2 += t[j] * t[j]; }
    float inv = 1.f / fmaxf(sqrtf(n2), 1e-12f);
#pragma unroll
    for (int j = 0; j < 5; ++j) U[(size_t)w * 5 + j] = t[j] * inv;
  }
}

// Attention score for (row, branch): tanh(s @ W1 + b1) @ W2 -- one wave each
__global__ __launch_bounds__(256)
void att_score_kernel(const float* __restrict__ Z, const float* __restrict__ ZT,
                      const float* __restrict__ W1, const float* __restrict__ B1,
                      const float* __restrict__ W2, float* __restrict__ S, int N) {
  int w = blockIdx.x * 8 + (threadIdx.x >> 5);
  int lane = threadIdx.x & 31;
  if (w >= 2 * N) return;
  int r = w >> 1, br = w & 1;
  const float* s = (br ? ZT : Z) + (size_t)r * 64;
  float score = 0.f;
  for (int j = lane; j < 128; j += 32) {
    float h = B1[j];
    for (int k = 0; k < 64; ++k) h += s[k] * W1[(size_t)k * 128 + j];
    score += tanhf(h) * W2[j];
  }
  score = wave_sum(score);
  if (lane == 0) S[(size_t)r * 2 + br] = score;
}

// beta = softmax over 2 branches; EMB = beta0*z + beta1*z_topo
__global__ void att_fuse_kernel(const float* __restrict__ S, const float* __restrict__ Z,
                                const float* __restrict__ ZT, float* __restrict__ EMB,
                                int N) {
  int i = blockIdx.x * blockDim.x + threadIdx.x;
  if (i >= N * 64) return;
  int r = i >> 6;
  float s0 = S[(size_t)r * 2], s1 = S[(size_t)r * 2 + 1];
  float m = fmaxf(s0, s1);
  float e0 = expf(s0 - m), e1 = expf(s1 - m);
  float inv = 1.f / (e0 + e1);
  EMB[i] = (e0 * inv) * Z[i] + (e1 * inv) * ZT[i];
}

// row-wise softmax over 64 columns -- one wave per row (2 cols/lane)
__global__ __launch_bounds__(256)
void softmax64_kernel(const float* __restrict__ X, float* __restrict__ Y, int N) {
  int w = blockIdx.x * 8 + (threadIdx.x >> 5);
  int lane = threadIdx.x & 31;
  if (w >= N) return;
  const float* xr = X + (size_t)w * 64;
  float a = xr[lane], b = xr[lane + 32];
  float m = wave_max(fmaxf(a, b));
  float ea = expf(a - m), eb = expf(b - m);
  float s = wave_sum(ea + eb);
  Y[(size_t)w * 64 + lane]      = ea / s;
  Y[(size_t)w * 64 + lane + 32] = eb / s;
}

// Student-t q (v=1 -> q_ij = 1/(1+||z_i-c_j||^2), row-normalized)
__global__ __launch_bounds__(256)
void q_kernel(const float* __restrict__ Z, const float* __restrict__ Cl,
              float* __restrict__ Q, int N) {
  int w = blockIdx.x * 8 + (threadIdx.x >> 5);
  int lane = threadIdx.x & 31;
  if (w >= N) return;
  const float* zr = Z + (size_t)w * 64;
  float z0 = zr[lane], z1 = zr[lane + 32];
  float zn = wave_sum(z0 * z0 + z1 * z1);
  float qv[2];
#pragma unroll
  for (int h = 0; h < 2; ++h) {
    int j = lane + 32 * h;
    const float* cj = Cl + (size_t)j * 64;
    float dot = 0.f, cn = 0.f;
    for (int k = 0; k < 64; ++k) { float c = cj[k]; dot += zr[k] * c; cn += c * c; }
    float d2 = zn + cn - 2.f * dot;
    qv[h] = 1.f / (1.f + d2);
  }
  float qs = wave_sum(qv[0] + qv[1]);
  Q[(size_t)w * 64 + lane]      = qv[0] / qs;
  Q[(size_t)w * 64 + lane + 32] = qv[1] / qs;
}

// ---------------------------------------------------------------------------
extern "C" void kernel_launch(void* const* d_in, const int* in_sizes, int n_in,
                              void* d_out, int out_size, void* d_ws, size_t ws_size,
                              hipStream_t stream) {
  const float* x       = (const float*)d_in[0];
  const float* adj_val = (const float*)d_in[1];
  const float* enc1_w  = (const float*)d_in[2];
  const float* enc1_b  = (const float*)d_in[3];
  const float* enc2_w  = (const float*)d_in[4];
  const float* enc2_b  = (const float*)d_in[5];
  const float* enc3_w  = (const float*)d_in[6];
  const float* enc3_b  = (const float*)d_in[7];
  const float* zl_w    = (const float*)d_in[8];
  const float* zl_b    = (const float*)d_in[9];
  const float* dec1_w  = (const float*)d_in[10];
  const float* dec1_b  = (const float*)d_in[11];
  const float* dec2_w  = (const float*)d_in[12];
  const float* dec2_b  = (const float*)d_in[13];
  const float* dec3_w  = (const float*)d_in[14];
  const float* dec3_b  = (const float*)d_in[15];
  const float* xbar_w  = (const float*)d_in[16];
  const float* xbar_b  = (const float*)d_in[17];
  const float* t1_w    = (const float*)d_in[18];
  const float* t2_w    = (const float*)d_in[19];
  const float* t3_w    = (const float*)d_in[20];
  const float* t4_w    = (const float*)d_in[21];
  const float* agcn_w  = (const float*)d_in[22];
  const float* mlpl_w  = (const float*)d_in[23];
  const float* mlpl_b  = (const float*)d_in[24];
  const float* att_w1  = (const float*)d_in[25];
  const float* att_b1  = (const float*)d_in[26];
  const float* att_w2  = (const float*)d_in[27];
  const float* cluster = (const float*)d_in[28];
  const int*   adj_row = (const int*)d_in[29];
  const int*   adj_col = (const int*)d_in[30];

  const int NIN = 2000;
  const int N = in_sizes[0] / NIN;      // 10000
  const int E = in_sizes[1];            // 100000
  const int NZ = 64;

  float* out   = (float*)d_out;
  float* xbar_o = out;                               // [N, 2000]
  float* q_o    = out + (size_t)N * NIN;             // [N, 64]
  float* pred_o = q_o + (size_t)N * NZ;              // [N, 64]
  float* emb_o  = pred_o + (size_t)N * NZ;           // [N, 64]

  // ---- workspace carve-up (~211 MB, heavily reused) ----
  float* ws = (float*)d_ws;
  size_t off = 0;
  float* bufA  = ws + off; off += (size_t)N * 2000;  // big temp (d1, tN-pre, agcn-pre)
  float* bufB  = ws + off; off += (size_t)N * 500;   // h1/d2 then T1 (persist)
  float* bufC  = ws + off; off += (size_t)N * 2000;  // h3 then T3 (persist)
  float* bufD  = ws + off; off += (size_t)N * 500;   // h2/d3 then T2 (persist)
  float* Zb    = ws + off; off += (size_t)N * NZ;    // z (persist)
  float* T4b   = ws + off; off += (size_t)N * NZ;    // t4 (persist)
  float* Ub    = ws + off; off += (size_t)N * 5;     // gate u
  float* ZTb   = ws + off; off += (size_t)N * NZ;    // z_topo
  float* EMBb  = ws + off; off += (size_t)N * NZ;    // pre-bn fused emb
  float* Sb    = ws + off; off += (size_t)N * 2;     // attention scores
  float* stats = ws + off; off += 4096;              // bn mean/invstd

  auto gemm = [&](const float* A, const float* B, const float* bias,
                  const float* rs, int rss, float* Cout,
                  int M, int K, int Nc, int act, int accum) {
    dim3 grid((Nc + GEMM_BN - 1) / GEMM_BN, (M + GEMM_BM - 1) / GEMM_BM);
    gemm_wmma_bf16<<<grid, 256, 0, stream>>>(A, B, bias, rs, rss, Cout,
                                             M, K, Nc, act, accum);
  };
  auto fill = [&](float* p, size_t n) {
    fill_zero<<<(unsigned)((n + 255) / 256), 256, 0, stream>>>(p, n);
  };
  auto spmm = [&](const float* H, float* O, int D) {
    fill(O, (size_t)N * D);
    spmm_kernel<<<E, 256, 0, stream>>>(adj_row, adj_col, adj_val, H, O, E, D);
  };
  auto act = [&](float* p, size_t n, int lrelu) {
    act_kernel<<<(unsigned)((n + 255) / 256), 256, 0, stream>>>(p, n, lrelu);
  };
  auto bn = [&](const float* in, float* outp, int D) {
    dim3 blk(32, 8);
    bn_stats_kernel<<<(D + 31) / 32, blk, 0, stream>>>(in, N, D, stats);
    size_t n = (size_t)N * D;
    bn_apply_kernel<<<(unsigned)((n + 255) / 256), 256, 0, stream>>>(in, outp, N, D, stats);
  };

  // ================= Autoencoder =================
  gemm(x,    enc1_w, enc1_b, nullptr, 0, bufB, N, 2000, 500, 1, 0);   // h1
  gemm(bufB, enc2_w, enc2_b, nullptr, 0, bufD, N,  500, 500, 1, 0);   // h2
  gemm(bufD, enc3_w, enc3_b, nullptr, 0, bufC, N,  500, 2000, 1, 0);  // h3
  gemm(bufC, zl_w,   zl_b,   nullptr, 0, Zb,   N, 2000,  NZ, 0, 0);   // z
  gemm(Zb,   dec1_w, dec1_b, nullptr, 0, bufA, N,   NZ, 2000, 1, 0);  // d1
  gemm(bufA, dec2_w, dec2_b, nullptr, 0, bufB, N, 2000, 500, 1, 0);   // d2
  gemm(bufB, dec3_w, dec3_b, nullptr, 0, bufD, N,  500, 500, 1, 0);   // d3
  gemm(bufD, xbar_w, xbar_b, nullptr, 0, xbar_o, N, 500, 2000, 0, 0); // x_bar

  // ================= Topo GCN stack =================
  float* T1b = bufB;   // persists
  float* T2b = bufD;   // persists
  float* T3b = bufC;   // persists
  gemm(x, t1_w, nullptr, nullptr, 0, bufA, N, 2000, 500, 0, 0);
  spmm(bufA, T1b, 500); act(T1b, (size_t)N * 500, 1); bn(T1b, T1b, 500);

  gemm(T1b, t2_w, nullptr, nullptr, 0, bufA, N, 500, 500, 0, 0);
  spmm(bufA, T2b, 500); act(T2b, (size_t)N * 500, 1); bn(T2b, T2b, 500);

  gemm(T2b, t3_w, nullptr, nullptr, 0, bufA, N, 500, 2000, 0, 0);
  spmm(bufA, T3b, 2000); act(T3b, (size_t)N * 2000, 1); bn(T3b, T3b, 2000);

  gemm(T3b, t4_w, nullptr, nullptr, 0, bufA, N, 2000, NZ, 0, 0);
  spmm(bufA, T4b, NZ); act(T4b, (size_t)N * NZ, 1); bn(T4b, T4b, NZ);

  // gate u over concat(t1..t4)
  u_kernel<<<(N + 7) / 8, 256, 0, stream>>>(T1b, T2b, T3b, T4b, mlpl_w, mlpl_b, Ub, N);

  // z_in @ agcn_w as 4 row-scaled partial GEMMs accumulated into bufA[:,0:64]
  gemm(T1b, agcn_w +    0 * 64, nullptr, Ub + 0, 5, bufA, N,  500, NZ, 0, 0);
  gemm(T2b, agcn_w +  500 * 64, nullptr, Ub + 1, 5, bufA, N,  500, NZ, 0, 1);
  gemm(T3b, agcn_w + 1000 * 64, nullptr, Ub + 2, 5, bufA, N, 2000, NZ, 0, 1);
  gemm(T4b, agcn_w + 3000 * 64, nullptr, Ub + 3, 5, bufA, N,   NZ, NZ, 0, 1);
  spmm(bufA, ZTb, NZ); act(ZTb, (size_t)N * NZ, 0); bn(ZTb, ZTb, NZ);  // z_topo

  // ================= Attention fusion =================
  att_score_kernel<<<(2 * N + 7) / 8, 256, 0, stream>>>(Zb, ZTb, att_w1, att_b1,
                                                        att_w2, Sb, N);
  att_fuse_kernel<<<(N * 64 + 255) / 256, 256, 0, stream>>>(Sb, Zb, ZTb, EMBb, N);
  bn(EMBb, emb_o, NZ);                                   // emb
  softmax64_kernel<<<(N + 7) / 8, 256, 0, stream>>>(emb_o, pred_o, N);  // pred

  // ================= Student-t q =================
  q_kernel<<<(N + 7) / 8, 256, 0, stream>>>(Zb, cluster, q_o, N);
}
